// GCNLayerFull_81080392614620
// MI455X (gfx1250) — matrix operand order repaired and checked by reading, not compiled, not run on hardware
//
#include <hip/hip_runtime.h>
#include <hip/hip_bf16.h>

// ---------------------------------------------------------------------------
// GCN layer: agg = segment_sum(features[src] -> dst); h = normalize(f + agg);
// out = h @ W^T + b.   N=100000, E=1600000, D_IN=D_OUT=128.
// ---------------------------------------------------------------------------

#define D 128
#define LDS_STRIDE_H 136   // bf16 elements per LDS row: 272B = 68 words -> 4-bank/row rotation

typedef __attribute__((ext_vector_type(16))) __bf16 v16bf;
typedef __attribute__((ext_vector_type(8)))  __bf16 v8bf;
typedef __attribute__((ext_vector_type(4)))  __bf16 v4bf;
typedef __attribute__((ext_vector_type(8)))  float  v8f;

union AFrag {
    v16bf v16;
    v8bf  v8[2];
};

// ---------------------------------------------------------------------------
// Kernel 1: zero the aggregation scratch (N*D floats) — required every call.
// ---------------------------------------------------------------------------
__global__ __launch_bounds__(256) void gcn_zero_agg(float* __restrict__ agg, long long n4)
{
    long long i = (long long)blockIdx.x * blockDim.x + threadIdx.x;
    if (i < n4) {
        ((float4*)agg)[i] = make_float4(0.f, 0.f, 0.f, 0.f);
    }
}

// ---------------------------------------------------------------------------
// Kernel 1b: convert W (128x128 fp32, row-major) to bf16 once per launch.
// ---------------------------------------------------------------------------
__global__ __launch_bounds__(256) void gcn_cvt_w(const float* __restrict__ W,
                                                 __bf16* __restrict__ Wb)
{
    int i = (blockIdx.x * blockDim.x + threadIdx.x) * 4;   // 4 elems / thread
    const float4 f = *(const float4*)(W + i);
    v4bf o;
    o[0] = (__bf16)f.x;
    o[1] = (__bf16)f.y;
    o[2] = (__bf16)f.z;
    o[3] = (__bf16)f.w;
    *(v4bf*)(Wb + i) = o;
}

// ---------------------------------------------------------------------------
// Kernel 2: edge scatter. One wave32 per edge: lane loads float4 of the src
// row (512B coalesced) and atomically adds into the dst row. agg is 51.2MB,
// L2-resident (192MB L2), so the fp32 RMWs resolve at L2 atomic units.
// ---------------------------------------------------------------------------
__global__ __launch_bounds__(256) void gcn_scatter_edges(
    const float* __restrict__ feat,
    const int*   __restrict__ esrc,
    const int*   __restrict__ edst,
    float*       __restrict__ agg,
    int n_edges)
{
    int gwave = (int)((blockIdx.x * blockDim.x + threadIdx.x) >> 5);
    int lane  = threadIdx.x & 31;
    if (gwave >= n_edges) return;

    int s = esrc[gwave];
    int d = edst[gwave];

    const float4 f = *(const float4*)(feat + (size_t)s * D + lane * 4);
    float* ap = agg + (size_t)d * D + lane * 4;
    atomicAdd(ap + 0, f.x);
    atomicAdd(ap + 1, f.y);
    atomicAdd(ap + 2, f.z);
    atomicAdd(ap + 3, f.w);
}

// ---------------------------------------------------------------------------
// Phase-1 helper: build one normalized bf16 row in LDS.
// ---------------------------------------------------------------------------
template <bool GUARDED>
__device__ __forceinline__ void gcn_row_to_lds(
    const float* __restrict__ feat, const float* __restrict__ agg,
    const bool* __restrict__ mask, __bf16* __restrict__ hs,
    int node, int r, int lane, int n_nodes)
{
    float h0 = 0.f, h1 = 0.f, h2 = 0.f, h3 = 0.f;
    if (!GUARDED || node < n_nodes) {
        const float4 f4 = *(const float4*)(feat + (size_t)node * D + lane * 4);
        const float4 a4 = *(const float4*)(agg  + (size_t)node * D + lane * 4);
        const bool m = mask[node];
        h0 = m ? (f4.x + a4.x) : f4.x;
        h1 = m ? (f4.y + a4.y) : f4.y;
        h2 = m ? (f4.z + a4.z) : f4.z;
        h3 = m ? (f4.w + a4.w) : f4.w;
    }
    float ss = h0 * h0 + h1 * h1 + h2 * h2 + h3 * h3;
    #pragma unroll
    for (int off = 16; off > 0; off >>= 1)
        ss += __shfl_xor(ss, off, 32);
    const float inv = 1.0f / fmaxf(sqrtf(ss), 1e-12f);
    v4bf o;
    o[0] = (__bf16)(h0 * inv);
    o[1] = (__bf16)(h1 * inv);
    o[2] = (__bf16)(h2 * inv);
    o[3] = (__bf16)(h3 * inv);
    *(v4bf*)(hs + r * LDS_STRIDE_H + lane * 4) = o;   // 8B aligned store
}

// ---------------------------------------------------------------------------
// Kernel 3: fused (features+agg) mask, L2-normalize, and 128x128 linear via
// v_wmma_f32_16x16x32_bf16.  Block = 256 threads = 8 waves = 16 nodes.
//   Phase 1: wave w computes h_norm rows 2w, 2w+1 -> LDS as bf16.
//   Phase 2: wave w computes the 16x16 output tile at columns [16w,16w+16):
//            bias preloaded into the accumulator (C(m,n)=bias[n] is constant
//            down columns), 4 chained K=32 WMMA steps, then 8 coalesced
//            immediate-offset stores (full-tile fast path; N % 16 == 0).
// ---------------------------------------------------------------------------
__global__ __launch_bounds__(256) void gcn_norm_gemm(
    const float*  __restrict__ feat,
    const float*  __restrict__ agg,
    const bool*   __restrict__ mask,
    const __bf16* __restrict__ Wb,    // [128][128] bf16 row-major
    const float*  __restrict__ bias,  // [128]
    float*        __restrict__ out,   // [N][128]
    int n_nodes)
{
    __shared__ __align__(16) __bf16 hs[16 * LDS_STRIDE_H];

    const int node_base = blockIdx.x * 16;
    const int wid  = threadIdx.x >> 5;   // 0..7
    const int lane = threadIdx.x & 31;   // 0..31
    const int nl   = lane & 15;          // lane % 16
    const int half = lane >> 4;          // 0 or 1
    const bool full = (node_base + 16 <= n_nodes);   // block-uniform

    // ---------------- Phase 1: normalized bf16 rows in LDS ----------------
    if (full) {
        #pragma unroll
        for (int rr = 0; rr < 2; ++rr) {
            const int r = 2 * wid + rr;
            gcn_row_to_lds<false>(feat, agg, mask, hs, node_base + r, r, lane, n_nodes);
        }
    } else {
        #pragma unroll
        for (int rr = 0; rr < 2; ++rr) {
            const int r = 2 * wid + rr;
            gcn_row_to_lds<true>(feat, agg, mask, hs, node_base + r, r, lane, n_nodes);
        }
    }
    __syncthreads();

    // ---------------- Phase 2: WMMA GEMM tile -------------------------------
    const int ct = wid;   // this wave's 16-column output tile

    // B fragments: contiguous K order, same as A: lanes 0-15 hold K=0..15, lanes
    // 16-31 hold K=16..31; N = lane&15; VGPR i packs K={2i,2i+1}. Element j of
    // v16bf = Wb[(16*ct+nl)*128 + 32*ks + 16*half + j] — two aligned b128 loads.
    const __bf16* wrow = Wb + (size_t)(16 * ct + nl) * D + 16 * half;
    AFrag bfrag[4];
    #pragma unroll
    for (int ks = 0; ks < 4; ++ks) {
        bfrag[ks].v8[0] = *(const v8bf*)(wrow + 32 * ks);
        bfrag[ks].v8[1] = *(const v8bf*)(wrow + 32 * ks + 8);
    }

    // Bias folded into accumulator: C(m, n) = bias[n] for every m.
    const float bv = bias[16 * ct + nl];
    v8f c;
    #pragma unroll
    for (int v = 0; v < 8; ++v) c[v] = bv;

    // A fragments from LDS, same K order as B per lane:
    //   elems 0..15 -> K = 32*ks + 16*half + (0..15)
    const __bf16* arow = hs + nl * LDS_STRIDE_H + 16 * half;

    #pragma unroll
    for (int ks = 0; ks < 4; ++ks) {
        AFrag a;
        a.v8[0] = *(const v8bf*)(arow + 32 * ks);
        a.v8[1] = *(const v8bf*)(arow + 32 * ks + 8);
        c = __builtin_amdgcn_wmma_f32_16x16x32_bf16(
                /*neg_a=*/false, a.v16,
                /*neg_b=*/false, bfrag[ks].v16,
                /*c_mod=*/(short)0, c,
                /*reuse_a=*/false, /*reuse_b=*/false);
    }

    // Store D tile. D layout: VGPR v, lanes 0-15 -> M=v, lanes 16-31 -> M=v+8;
    // N = lane&15.  Row stride between consecutive VGPRs is D floats (512B),
    // expressible as an immediate offset -> one address + 8 coalesced stores.
    float* obase = out + (size_t)(node_base + 8 * half) * D + 16 * ct + nl;
    if (full) {
        #pragma unroll
        for (int v = 0; v < 8; ++v)
            obase[(size_t)v * D] = c[v];
    } else {
        #pragma unroll
        for (int v = 0; v < 8; ++v) {
            if (node_base + 8 * half + v < n_nodes)
                obase[(size_t)v * D] = c[v];
        }
    }
}

// ---------------------------------------------------------------------------
// Host-side launcher (graph-capture safe: kernels only, all on `stream`).
// ---------------------------------------------------------------------------
extern "C" void kernel_launch(void* const* d_in, const int* in_sizes, int n_in,
                              void* d_out, int out_size, void* d_ws, size_t ws_size,
                              hipStream_t stream)
{
    const float* feat = (const float*)d_in[0];
    const int*   esrc = (const int*)  d_in[1];
    const int*   edst = (const int*)  d_in[2];
    const bool*  mask = (const bool*) d_in[3];
    const float* W    = (const float*)d_in[4];
    const float* bias = (const float*)d_in[5];
    float*       out  = (float*)d_out;

    const int n_nodes = in_sizes[0] / D;   // 100000
    const int n_edges = in_sizes[1];       // 1600000

    // Workspace layout: [ agg: N*D fp32 | Wb: 128*128 bf16 ]
    float*  agg = (float*)d_ws;
    __bf16* Wb  = (__bf16*)((char*)d_ws + (size_t)n_nodes * D * sizeof(float));

    // 1) zero agg
    {
        long long n4 = (long long)n_nodes * D / 4;
        int blocks = (int)((n4 + 255) / 256);
        gcn_zero_agg<<<blocks, 256, 0, stream>>>(agg, n4);
    }
    // 1b) pre-convert W to bf16 (16384 elems, 4 per thread)
    {
        gcn_cvt_w<<<(D * D / 4 + 255) / 256, 256, 0, stream>>>(W, Wb);
    }
    // 2) edge scatter (one wave32 per edge; 8 edges per 256-thread block)
    {
        int blocks = (n_edges + 7) / 8;
        gcn_scatter_edges<<<blocks, 256, 0, stream>>>(feat, esrc, edst, agg, n_edges);
    }
    // 3) fused normalize + WMMA linear (16 nodes per block)
    {
        int blocks = (n_nodes + 15) / 16;
        gcn_norm_gemm<<<blocks, 256, 0, stream>>>(feat, agg, mask, Wb, bias, out, n_nodes);
    }
}
